// CrossTokenSelectorSaliency_new_13073880449115
// MI455X (gfx1250) — matrix operand order repaired and checked by reading, not compiled, not run on hardware
//
#include <hip/hip_runtime.h>
#include <hip/hip_bf16.h>
#include <math.h>

// ---------------------------------------------------------------------------
// CDNA5 (gfx1250) implementation of CrossTokenSelectorSaliency forward.
// Heavy matmuls: v_wmma_f32_16x16x32_f16 (f16 in, f32 accumulate), 32x32
// wave tiles (4 WMMAs per fragment set). Tile staging uses the CDNA5 async
// global->LDS path (global_load_async_to_lds_b128 + s_wait_asynccnt).
// Workspace requirement: ~330 MB (bump-allocated below).
// ---------------------------------------------------------------------------

#define B_    16
#define NB    1024
#define NS    4096
#define CH    768
#define C4    3072
#define NHEAD 12
#define DH    64
#define TOPK  128

typedef __attribute__((ext_vector_type(16))) _Float16 v16h;
typedef __attribute__((ext_vector_type(8)))  float    v8f;

#define DEVINL __device__ __forceinline__

// ---- CDNA5 async global->LDS copy (ASYNCcnt-tracked) ----------------------
// LDS dest address = low 32 bits of generic shared pointer (LDS aperture).

DEVINL unsigned lds_addr(const void* p) {
  return (unsigned)(size_t)p;
}

DEVINL void async_ld_b128(unsigned lds, const void* gaddr) {
  asm volatile("global_load_async_to_lds_b128 %0, %1, off"
               :
               : "v"(lds), "v"(gaddr)
               : "memory");
}

DEVINL void wait_async0() {
  asm volatile("s_wait_asynccnt 0x0" ::: "memory");
}

// ---- WMMA fragment loaders (CDNA5 wave32 layouts, 16x16x32 f16) -----------

// A-matrix 16x32 f16 from LDS tile stored row-major [16+][ldm].
// lane 0-15: M=lane, K = 0..15 in elems 0-15
// lane 16-31: M=lane-16, K = 16..31
DEVINL v16h frag_a_ld(const _Float16* p, int ldm) {
  const int lane = threadIdx.x & 31;
  const int m = lane & 15;
  const int kb = (lane < 16) ? 0 : 16;
  v16h f;
#pragma unroll
  for (int j = 0; j < 16; ++j) {
    f[j] = p[m * ldm + kb + j];
  }
  return f;
}

// B-matrix 32x16 f16 from LDS tile stored row-major [K][ldn] (K outer).
// lane 0-15: N=lane, K=0..15 ; lane 16-31: N=lane-16, K=16..31
DEVINL v16h frag_b_ld(const _Float16* p, int ldn) {
  const int lane = threadIdx.x & 31;
  const int n = lane & 15;
  const int kg = (lane < 16) ? 0 : 16;
  v16h f;
#pragma unroll
  for (int j = 0; j < 16; ++j) f[j] = p[(kg + j) * ldn + n];
  return f;
}

// B-matrix 32x16 f16 from LDS tile stored TRANSPOSED: [N][ldk].
DEVINL v16h frag_bt_ld(const _Float16* p, int ldk) {
  const int lane = threadIdx.x & 31;
  const int n = lane & 15;
  const int kg = (lane < 16) ? 0 : 16;
  v16h f;
#pragma unroll
  for (int j = 0; j < 16; ++j) f[j] = p[n * ldk + kg + j];
  return f;
}

DEVINL v8f wmma32(v16h a, v16h b, v8f c) {
  return __builtin_amdgcn_wmma_f32_16x16x32_f16(false, a, false, b,
                                                (short)0, c, false, false);
}

// ---- fp32 -> fp16 conversion ----------------------------------------------

__global__ void cvt_f32_f16(const float* __restrict__ in,
                            _Float16* __restrict__ out, long n) {
  long i = (long)blockIdx.x * blockDim.x + threadIdx.x;
  long stride = (long)gridDim.x * blockDim.x;
  for (long j = i * 4; j < n; j += stride * 4) {
    float4 v = *(const float4*)&in[j];
    out[j + 0] = (_Float16)v.x;
    out[j + 1] = (_Float16)v.y;
    out[j + 2] = (_Float16)v.z;
    out[j + 3] = (_Float16)v.w;
  }
}

// ---- saliency: rowmax over (sampled . base^T) / sqrt(C) -------------------
// grid = (NS/128, B), block = 256 (8 waves, 4x2 wave grid, wave tile 32x32)

__global__ void __launch_bounds__(256)
saliency_kernel(const _Float16* __restrict__ S,   // [B,NS,C] f16
                const _Float16* __restrict__ Bt,  // [B,NB,C] f16
                float* __restrict__ scores) {     // [B,NS]
  __shared__ _Float16 As[128 * 40];
  __shared__ _Float16 Bs[64 * 40];
  __shared__ float red[2][128];

  const int tid = threadIdx.x;
  const int wave = tid >> 5, wm = wave & 3, wn = wave >> 2;
  const int b = blockIdx.y;
  const int m0 = blockIdx.x * 128;
  const size_t sbase = (size_t)b * NS * CH;
  const size_t bbase = (size_t)b * NB * CH;
  const int arow = tid >> 2, ac8 = tid & 3;  // 2 A tasks + 1 B task / thread
  const unsigned ldsA0 = lds_addr(&As[arow * 40 + ac8 * 8]);
  const unsigned ldsA1 = lds_addr(&As[(arow + 64) * 40 + ac8 * 8]);
  const unsigned ldsB  = lds_addr(&Bs[arow * 40 + ac8 * 8]);

  float rm[2][2][8];
#pragma unroll
  for (int mi = 0; mi < 2; ++mi)
#pragma unroll
    for (int t = 0; t < 2; ++t)
#pragma unroll
      for (int r = 0; r < 8; ++r) rm[mi][t][r] = -3.4e38f;

  for (int nt = 0; nt < NB / 64; ++nt) {
    v8f acc[2][2];
#pragma unroll
    for (int mi = 0; mi < 2; ++mi)
#pragma unroll
      for (int t = 0; t < 2; ++t)
#pragma unroll
        for (int r = 0; r < 8; ++r) acc[mi][t][r] = 0.0f;

    for (int ks = 0; ks < CH / 32; ++ks) {
      const int k0 = ks << 5;
      __syncthreads();
      async_ld_b128(ldsA0, &S[sbase + (size_t)(m0 + arow) * CH + k0 + ac8 * 8]);
      async_ld_b128(ldsA1,
                    &S[sbase + (size_t)(m0 + arow + 64) * CH + k0 + ac8 * 8]);
      async_ld_b128(ldsB,
                    &Bt[bbase + (size_t)(nt * 64 + arow) * CH + k0 + ac8 * 8]);
      wait_async0();
      __syncthreads();
      v16h af0 = frag_a_ld(&As[(wm * 32) * 40], 40);
      v16h af1 = frag_a_ld(&As[(wm * 32 + 16) * 40], 40);
#pragma unroll
      for (int t = 0; t < 2; ++t) {
        v16h bf = frag_bt_ld(&Bs[(wn * 32 + t * 16) * 40], 40);
        acc[0][t] = wmma32(af0, bf, acc[0][t]);
        acc[1][t] = wmma32(af1, bf, acc[1][t]);
      }
    }
#pragma unroll
    for (int mi = 0; mi < 2; ++mi)
#pragma unroll
      for (int t = 0; t < 2; ++t)
#pragma unroll
        for (int r = 0; r < 8; ++r)
          rm[mi][t][r] = fmaxf(rm[mi][t][r], acc[mi][t][r]);
  }

  const int lane = tid & 31;
#pragma unroll
  for (int mi = 0; mi < 2; ++mi)
#pragma unroll
    for (int r = 0; r < 8; ++r) {
      float v = fmaxf(rm[mi][0][r], rm[mi][1][r]);
      v = fmaxf(v, __shfl_xor(v, 1, 32));
      v = fmaxf(v, __shfl_xor(v, 2, 32));
      v = fmaxf(v, __shfl_xor(v, 4, 32));
      v = fmaxf(v, __shfl_xor(v, 8, 32));
      if ((lane & 15) == 0)
        red[wn][wm * 32 + mi * 16 + (lane >> 4) * 8 + r] = v;
    }
  __syncthreads();
  if (tid < 128) {
    const float scale = 0.03608439182435161f;  // 1/sqrt(768)
    scores[(size_t)b * NS + m0 + tid] =
        fmaxf(red[0][tid], red[1][tid]) * scale;
  }
}

// ---- top-k (sorted descending, lower-index ties) + gather -----------------
// grid = B, block = 256

__global__ void __launch_bounds__(256)
topk_gather_kernel(const float* __restrict__ scores,   // [B,NS]
                   const float* __restrict__ sampled,  // [B,NS,C] f32
                   int* __restrict__ topkOut,          // [B,TOPK]
                   float* __restrict__ selF32,         // [B,TOPK,C]
                   _Float16* __restrict__ selF16) {    // [B,TOPK,C]
  __shared__ float sv[NS];
  __shared__ float bval[256];
  __shared__ int bidx[256];
  __shared__ int sel[TOPK];
  const int b = blockIdx.x, tid = threadIdx.x;

  for (int i = tid; i < NS; i += 256) sv[i] = scores[(size_t)b * NS + i];
  __syncthreads();

  for (int k = 0; k < TOPK; ++k) {
    float best = -3.4e38f;
    int bi = NS;
    for (int i = tid; i < NS; i += 256) {
      float v = sv[i];
      if (v > best || (v == best && i < bi)) { best = v; bi = i; }
    }
    bval[tid] = best; bidx[tid] = bi;
    __syncthreads();
    for (int s = 128; s > 0; s >>= 1) {
      if (tid < s) {
        float ov = bval[tid + s]; int oi = bidx[tid + s];
        if (ov > bval[tid] || (ov == bval[tid] && oi < bidx[tid])) {
          bval[tid] = ov; bidx[tid] = oi;
        }
      }
      __syncthreads();
    }
    if (tid == 0) { sel[k] = bidx[0]; sv[bidx[0]] = -3.4e38f; }
    __syncthreads();
  }

  if (tid < TOPK) topkOut[b * TOPK + tid] = sel[tid];
  for (int k = 0; k < TOPK; ++k) {
    const int rs = sel[k];
    const size_t src = ((size_t)b * NS + rs) * CH;
    const size_t dst = ((size_t)b * TOPK + k) * CH;
    for (int c = tid; c < CH; c += 256) {
      float v = sampled[src + c];
      selF32[dst + c] = v;
      selF16[dst + c] = (_Float16)v;
    }
  }
}

// ---- generic WMMA GEMM: out[M,N] = A16[M,Ka] * W16[Ka,N] + bias -----------
// EPI 0: f16 out (bias)          -- Q/K/V projections
// EPI 1: f32 out (bias+residual) -- Wo, Wm2
// EPI 2: f16 out (bias+GELU)     -- Wm1
// grid = (N/64, M/128), block = 256; wave tile 32x32 (4 WMMAs / frag set)

template <int EPI>
__global__ void __launch_bounds__(256)
wmma_gemm_kernel(const _Float16* __restrict__ A, const _Float16* __restrict__ W,
                 const float* __restrict__ bias,
                 const float* __restrict__ residual,
                 float* __restrict__ outF32, _Float16* __restrict__ outF16,
                 int M, int N, int Ka) {
  __shared__ _Float16 As[128 * 40];
  __shared__ _Float16 Bs[32 * 72];
  const int tid = threadIdx.x;
  const int wave = tid >> 5, wm = wave & 3, wn = wave >> 2;
  const int m0 = blockIdx.y * 128;
  const int n0 = blockIdx.x * 64;

  v8f acc[2][2];
#pragma unroll
  for (int mi = 0; mi < 2; ++mi)
#pragma unroll
    for (int t = 0; t < 2; ++t)
#pragma unroll
      for (int r = 0; r < 8; ++r) acc[mi][t][r] = 0.0f;

  const int arow = tid >> 2, ac8 = tid & 3;  // A tile 128x32: 2 tasks/thread
  const int bkr = tid >> 3, bnc = tid & 7;   // B tile 32x64: 1 task/thread
  const unsigned ldsA0 = lds_addr(&As[arow * 40 + ac8 * 8]);
  const unsigned ldsA1 = lds_addr(&As[(arow + 64) * 40 + ac8 * 8]);
  const unsigned ldsB  = lds_addr(&Bs[bkr * 72 + bnc * 8]);

  const int nsteps = Ka >> 5;
  for (int ks = 0; ks < nsteps; ++ks) {
    const int k0 = ks << 5;
    __syncthreads();
    async_ld_b128(ldsA0, &A[(size_t)(m0 + arow) * Ka + k0 + ac8 * 8]);
    async_ld_b128(ldsA1, &A[(size_t)(m0 + arow + 64) * Ka + k0 + ac8 * 8]);
    async_ld_b128(ldsB, &W[(size_t)(k0 + bkr) * N + n0 + bnc * 8]);
    wait_async0();
    __syncthreads();
    v16h af0 = frag_a_ld(&As[(wm * 32) * 40], 40);
    v16h af1 = frag_a_ld(&As[(wm * 32 + 16) * 40], 40);
#pragma unroll
    for (int t = 0; t < 2; ++t) {
      v16h bf = frag_b_ld(&Bs[wn * 32 + t * 16], 72);
      acc[0][t] = wmma32(af0, bf, acc[0][t]);
      acc[1][t] = wmma32(af1, bf, acc[1][t]);
    }
  }

  const int lane = tid & 31;
  const int hs = lane >> 4, nn = lane & 15;
#pragma unroll
  for (int mi = 0; mi < 2; ++mi)
#pragma unroll
    for (int t = 0; t < 2; ++t) {
      const int col = n0 + wn * 32 + t * 16 + nn;
      const float bb = bias[col];
#pragma unroll
      for (int r = 0; r < 8; ++r) {
        const int row = m0 + wm * 32 + mi * 16 + hs * 8 + r;
        const size_t idx = (size_t)row * N + col;
        float v = acc[mi][t][r] + bb;
        if (EPI == 0) {
          outF16[idx] = (_Float16)v;
        } else if (EPI == 1) {
          outF32[idx] = v + residual[idx];
        } else {
          v = 0.5f * v * (1.0f + erff(v * 0.70710678118f));  // exact GELU
          outF16[idx] = (_Float16)v;
        }
      }
    }
}

// ---- cross attention: logits -> softmax -> attn*V -------------------------
// grid = (NB/64, H, B), block = 256, dynamic LDS

#define ATTN_SMEM ((64 * 72 + 128 * 72 + 128 * 72) * 2 + 64 * 132 * 4 + 64 * 136 * 2)

__global__ void __launch_bounds__(256)
attention_kernel(const _Float16* __restrict__ Q,  // [B,NB,C]
                 const _Float16* __restrict__ K,  // [B,TOPK,C]
                 const _Float16* __restrict__ V,  // [B,TOPK,C]
                 float* __restrict__ attnOut,     // [B,H,NB,TOPK]
                 _Float16* __restrict__ attended) {  // [B,NB,C]
  extern __shared__ char smem[];
  _Float16* Qs = (_Float16*)smem;              // 64 x 72
  _Float16* Ks = Qs + 64 * 72;                 // 128 x 72
  _Float16* Vs = Ks + 128 * 72;                // 128 x 72
  float* Lg = (float*)(Vs + 128 * 72);         // 64 x 132 (logits / exp)
  _Float16* Ps = (_Float16*)(Lg + 64 * 132);   // 64 x 136 (f16 probs)

  const int tid = threadIdx.x;
  const int q0 = blockIdx.x * 64;
  const int h = blockIdx.y;
  const int b = blockIdx.z;
  const size_t qbase = ((size_t)b * NB + q0) * CH + h * DH;
  const size_t kvbase = ((size_t)b * TOPK) * CH + h * DH;

  // async-stage Q (64x64), K,V (128x64) tiles
  const int lrow = tid >> 3, lc8 = tid & 7;
  async_ld_b128(lds_addr(&Qs[lrow * 72 + lc8 * 8]),
                &Q[qbase + (size_t)lrow * CH + lc8 * 8]);
  async_ld_b128(lds_addr(&Qs[(lrow + 32) * 72 + lc8 * 8]),
                &Q[qbase + (size_t)(lrow + 32) * CH + lc8 * 8]);
#pragma unroll
  for (int rr = 0; rr < 128; rr += 32) {
    async_ld_b128(lds_addr(&Ks[(lrow + rr) * 72 + lc8 * 8]),
                  &K[kvbase + (size_t)(lrow + rr) * CH + lc8 * 8]);
    async_ld_b128(lds_addr(&Vs[(lrow + rr) * 72 + lc8 * 8]),
                  &V[kvbase + (size_t)(lrow + rr) * CH + lc8 * 8]);
  }
  wait_async0();
  __syncthreads();

  const int wave = tid >> 5, wm = wave & 3, wn = wave >> 2;
  const int lane = tid & 31, hs = lane >> 4, nn = lane & 15;

  // logits 64x128; each wave: 16 rows x 64 cols (4 frags), Dh=64 -> 2 K-steps
  v8f lc[4];
#pragma unroll
  for (int f = 0; f < 4; ++f)
#pragma unroll
    for (int r = 0; r < 8; ++r) lc[f][r] = 0.0f;
  for (int ks = 0; ks < 2; ++ks) {
    v16h af = frag_a_ld(&Qs[(wm * 16) * 72 + ks * 32], 72);
#pragma unroll
    for (int f = 0; f < 4; ++f) {
      v16h bf = frag_bt_ld(&Ks[(wn * 64 + f * 16) * 72 + ks * 32], 72);
      lc[f] = wmma32(af, bf, lc[f]);
    }
  }
#pragma unroll
  for (int f = 0; f < 4; ++f) {
    const int col = wn * 64 + f * 16 + nn;
#pragma unroll
    for (int r = 0; r < 8; ++r)
      Lg[(wm * 16 + hs * 8 + r) * 132 + col] = lc[f][r] * 0.125f;  // 1/sqrt(64)
  }
  __syncthreads();

  // softmax: one thread per row
  if (tid < 64) {
    float mx = -3.4e38f;
    for (int c = 0; c < TOPK; ++c) mx = fmaxf(mx, Lg[tid * 132 + c]);
    float sum = 0.f;
    for (int c = 0; c < TOPK; ++c) {
      float e = __expf(Lg[tid * 132 + c] - mx);
      Lg[tid * 132 + c] = e;
      sum += e;
    }
    const float inv = 1.0f / sum;
    const size_t ob = (((size_t)b * NHEAD + h) * NB + q0 + tid) * TOPK;
    for (int c = 0; c < TOPK; ++c) {
      float p = Lg[tid * 132 + c] * inv;
      attnOut[ob + c] = p;
      Ps[tid * 136 + c] = (_Float16)p;
    }
  }
  __syncthreads();

  // attended 64x64 = P(64x128) * V(128x64); each wave 16x32, 4 K-steps
  v8f oc[2];
#pragma unroll
  for (int t = 0; t < 2; ++t)
#pragma unroll
    for (int r = 0; r < 8; ++r) oc[t][r] = 0.0f;
  for (int ks = 0; ks < 4; ++ks) {
    v16h af = frag_a_ld(&Ps[(wm * 16) * 136 + ks * 32], 136);
#pragma unroll
    for (int t = 0; t < 2; ++t) {
      v16h bf = frag_b_ld(&Vs[ks * 32 * 72 + wn * 32 + t * 16], 72);
      oc[t] = wmma32(af, bf, oc[t]);
    }
  }
#pragma unroll
  for (int t = 0; t < 2; ++t) {
    const int col = wn * 32 + t * 16 + nn;
#pragma unroll
    for (int r = 0; r < 8; ++r) {
      const int row = wm * 16 + hs * 8 + r;
      attended[((size_t)b * NB + q0 + row) * CH + h * DH + col] =
          (_Float16)oc[t][r];
    }
  }
}

// ---- fused LayerNorm (f32 out, optional f16 copy) -------------------------
// grid = B*NB, block = 256

__global__ void __launch_bounds__(256)
layernorm_kernel(const float* __restrict__ x, const float* __restrict__ g,
                 const float* __restrict__ bta, float* __restrict__ outF32,
                 _Float16* __restrict__ outF16) {
  const size_t row = blockIdx.x;
  const int tid = threadIdx.x;
  const float* xr = x + row * CH;
  float s = 0.f, sq = 0.f;
  for (int c = tid; c < CH; c += 256) {
    float v = xr[c];
    s += v;
    sq += v * v;
  }
#pragma unroll
  for (int m = 16; m > 0; m >>= 1) {
    s += __shfl_xor(s, m, 32);
    sq += __shfl_xor(sq, m, 32);
  }
  __shared__ float wsum[8], wsq[8];
  __shared__ float smu, srs;
  const int wave = tid >> 5;
  if ((tid & 31) == 0) { wsum[wave] = s; wsq[wave] = sq; }
  __syncthreads();
  if (tid == 0) {
    float ts = 0.f, tq = 0.f;
    for (int w = 0; w < 8; ++w) { ts += wsum[w]; tq += wsq[w]; }
    float mu = ts / (float)CH;
    smu = mu;
    srs = rsqrtf(tq / (float)CH - mu * mu + 1e-5f);
  }
  __syncthreads();
  for (int c = tid; c < CH; c += 256) {
    float v = (xr[c] - smu) * srs * g[c] + bta[c];
    outF32[row * CH + c] = v;
    if (outF16) outF16[row * CH + c] = (_Float16)v;
  }
}

// ---------------------------------------------------------------------------

extern "C" void kernel_launch(void* const* d_in, const int* in_sizes, int n_in,
                              void* d_out, int out_size, void* d_ws,
                              size_t ws_size, hipStream_t stream) {
  (void)in_sizes; (void)n_in; (void)out_size; (void)ws_size;
  const float* base    = (const float*)d_in[0];
  const float* sampled = (const float*)d_in[1];
  const float* Wq  = (const float*)d_in[2];
  const float* bq  = (const float*)d_in[3];
  const float* Wk  = (const float*)d_in[4];
  const float* bk  = (const float*)d_in[5];
  const float* Wv  = (const float*)d_in[6];
  const float* bv  = (const float*)d_in[7];
  const float* Wo  = (const float*)d_in[8];
  const float* bo  = (const float*)d_in[9];
  const float* g1  = (const float*)d_in[10];
  const float* b1  = (const float*)d_in[11];
  const float* g2  = (const float*)d_in[12];
  const float* b2  = (const float*)d_in[13];
  const float* Wm1 = (const float*)d_in[14];
  const float* bm1 = (const float*)d_in[15];
  const float* Wm2 = (const float*)d_in[16];
  const float* bm2 = (const float*)d_in[17];

  // d_out layout: x | sampled_scores | attn | topk_idx(int32) | selected
  float* out = (float*)d_out;
  float* outX      = out;
  float* outScores = out + (size_t)12582912;
  float* outAttn   = out + (size_t)12648448;
  int*   outTopk   = (int*)(out + (size_t)37814272);
  float* outSel    = out + (size_t)37816320;

  // workspace bump allocator (~330 MB total)
  char* wsb = (char*)d_ws;
  size_t off = 0;
  auto alloc = [&](size_t bytes) -> void* {
    void* p = wsb + off;
    off = (off + bytes + 255) & ~(size_t)255;
    return p;
  };
  _Float16* base16    = (_Float16*)alloc((size_t)B_ * NB * CH * 2);
  _Float16* sampled16 = (_Float16*)alloc((size_t)B_ * NS * CH * 2);  // reused as h16
  _Float16* Wq16  = (_Float16*)alloc((size_t)CH * CH * 2);
  _Float16* Wk16  = (_Float16*)alloc((size_t)CH * CH * 2);
  _Float16* Wv16  = (_Float16*)alloc((size_t)CH * CH * 2);
  _Float16* Wo16  = (_Float16*)alloc((size_t)CH * CH * 2);
  _Float16* Wm116 = (_Float16*)alloc((size_t)CH * C4 * 2);
  _Float16* Wm216 = (_Float16*)alloc((size_t)C4 * CH * 2);
  _Float16* sel16 = (_Float16*)alloc((size_t)B_ * TOPK * CH * 2);
  _Float16* q16   = (_Float16*)alloc((size_t)B_ * NB * CH * 2);
  _Float16* k16   = (_Float16*)alloc((size_t)B_ * TOPK * CH * 2);
  _Float16* v16   = (_Float16*)alloc((size_t)B_ * TOPK * CH * 2);
  _Float16* att16 = (_Float16*)alloc((size_t)B_ * NB * CH * 2);
  float*    y1    = (float*)alloc((size_t)B_ * NB * CH * 4);  // reused as y2
  float*    x1    = (float*)alloc((size_t)B_ * NB * CH * 4);
  _Float16* x1_16 = (_Float16*)alloc((size_t)B_ * NB * CH * 2);
  _Float16* h16   = sampled16;  // alias: free after top-k gather

  auto cvt = [&](const float* src, _Float16* dst, long n) {
    long blocks = (n / 4 + 255) / 256;
    if (blocks > 4096) blocks = 4096;
    cvt_f32_f16<<<(int)blocks, 256, 0, stream>>>(src, dst, n);
  };

  // 1) precision demotion for WMMA operands
  cvt(base,    base16,    (long)B_ * NB * CH);
  cvt(sampled, sampled16, (long)B_ * NS * CH);
  cvt(Wq,  Wq16,  (long)CH * CH);
  cvt(Wk,  Wk16,  (long)CH * CH);
  cvt(Wv,  Wv16,  (long)CH * CH);
  cvt(Wo,  Wo16,  (long)CH * CH);
  cvt(Wm1, Wm116, (long)CH * C4);
  cvt(Wm2, Wm216, (long)C4 * CH);

  // 2) saliency rowmax
  saliency_kernel<<<dim3(NS / 128, B_), 256, 0, stream>>>(sampled16, base16,
                                                          outScores);
  // 3) top-k + gather (sampled16 free afterwards -> reused as h16)
  topk_gather_kernel<<<B_, 256, 0, stream>>>(outScores, sampled, outTopk,
                                             outSel, sel16);
  // 4) projections
  wmma_gemm_kernel<0><<<dim3(CH / 64, (B_ * NB) / 128), 256, 0, stream>>>(
      base16, Wq16, bq, nullptr, nullptr, q16, B_ * NB, CH, CH);
  wmma_gemm_kernel<0><<<dim3(CH / 64, (B_ * TOPK) / 128), 256, 0, stream>>>(
      sel16, Wk16, bk, nullptr, nullptr, k16, B_ * TOPK, CH, CH);
  wmma_gemm_kernel<0><<<dim3(CH / 64, (B_ * TOPK) / 128), 256, 0, stream>>>(
      sel16, Wv16, bv, nullptr, nullptr, v16, B_ * TOPK, CH, CH);
  // 5) attention
  attention_kernel<<<dim3(NB / 64, NHEAD, B_), 256, ATTN_SMEM, stream>>>(
      q16, k16, v16, outAttn, att16);
  // 6) output projection + residual
  wmma_gemm_kernel<1><<<dim3(CH / 64, (B_ * NB) / 128), 256, 0, stream>>>(
      att16, Wo16, bo, base, y1, nullptr, B_ * NB, CH, CH);
  // 7) LN1 (f32 + f16 copy for next GEMM)
  layernorm_kernel<<<B_ * NB, 256, 0, stream>>>(y1, g1, b1, x1, x1_16);
  // 8) MLP up + GELU
  wmma_gemm_kernel<2><<<dim3(C4 / 64, (B_ * NB) / 128), 256, 0, stream>>>(
      x1_16, Wm116, bm1, nullptr, nullptr, h16, B_ * NB, C4, CH);
  // 9) MLP down + residual (y1 reused as y2)
  wmma_gemm_kernel<1><<<dim3(CH / 64, (B_ * NB) / 128), 256, 0, stream>>>(
      h16, Wm216, bm2, x1, y1, nullptr, B_ * NB, CH, C4);
  // 10) LN2 -> final x
  layernorm_kernel<<<B_ * NB, 256, 0, stream>>>(y1, g2, b2, outX, nullptr);
}
